// EGNN_11330123727315
// MI455X (gfx1250) — compile-verified, matrix operand order checked by reading, not run on hardware
//
#include <hip/hip_runtime.h>
#include <hip/hip_bf16.h>

// ---------------------------------------------------------------------------
// EGNN layer fused for MI455X (gfx1250, wave32, WMMA).
//   B=4 N=2048 D=128 K=32 FF=4 MDIM=64 EIN=265
// All GEMMs run on v_wmma_f32_16x16x32_f16 (f16 A/B, f32 accum).
// Edge pipeline fully fused in LDS (no edge_in / m_ij in HBM).
// ---------------------------------------------------------------------------

typedef __attribute__((ext_vector_type(16))) _Float16 v16h;
typedef __attribute__((ext_vector_type(8)))  float    v8f;
typedef __attribute__((ext_vector_type(4)))  unsigned int u4;

#define BN    8192      // B*N
#define NN    2048      // N
#define DD    128
#define KK    32
#define EINP  288       // 265 padded to 9*32
#define H1P   544       // 530 padded to 34*16 (= 17*32)
#define MD    64
#define C1    256
#define NINP  192       // 128+64
#define NH    256

__device__ __forceinline__ float silu_f(float x) {
  return x / (1.f + __expf(-x));
}

__device__ __forceinline__ v8f wmma_f16(v16h a, v16h b, v8f c) {
  return __builtin_amdgcn_wmma_f32_16x16x32_f16(false, a, false, b, (short)0, c,
                                                false, false);
}

// A fragment (16x32 f16) from a row-major LDS tile, row stride ld halves.
// Lane L<16 holds M=L, K in {kb..kb+7, kb+16..kb+23} with kb=0; L>=16: kb=8.
__device__ __forceinline__ v16h fragA_lds(const _Float16* base, int lane,
                                          int mA, int k0, int ld) {
  const _Float16* p = base + mA * ld + k0 + ((lane >> 4) << 3);
  v16h r;
  *((u4*)&r)     = *(const u4*)(p);
  *((u4*)&r + 1) = *(const u4*)(p + 16);
  return r;
}

// B fragment (32x16 f16) from prepacked weights: fragment f = kc*Nt+nt is
// 512 halves, lane-major (lane*16 .. lane*16+15).
__device__ __forceinline__ v16h fragB(const _Float16* Wp, int lane, int kc,
                                      int nt, int Nt) {
  const _Float16* p = Wp + (((size_t)(kc * Nt + nt)) << 9) + (lane << 4);
  v16h r;
  *((u4*)&r)     = *(const u4*)(p);
  *((u4*)&r + 1) = *(const u4*)(p + 8);
  return r;
}

// ---------------------------------------------------------------------------
// Prep: f32 -> f16 feats
__global__ __launch_bounds__(256) void k_feats16(const float* __restrict__ src,
                                                 _Float16* __restrict__ dst,
                                                 int n) {
  int i = blockIdx.x * 256 + threadIdx.x;
  if (i < n) dst[i] = (_Float16)src[i];
}

// Prep: pack f32 weight [Krows x Ncols] into zero-padded f16 WMMA fragments.
__global__ __launch_bounds__(256) void k_prepack(const float* __restrict__ src,
                                                 _Float16* __restrict__ dst,
                                                 int Krows, int Ncols, int Nt,
                                                 int total) {
  int i = blockIdx.x * 256 + threadIdx.x;
  if (i >= total) return;
  int f = i >> 9, w = i & 511;
  int lane = w >> 4, h = w & 15;
  int kc = f / Nt, nt = f - kc * Nt;
  int k = kc * 32 + ((lane >> 4) << 3) + (h < 8 ? h : h + 8);
  int n = nt * 16 + (lane & 15);
  float v = (k < Krows && n < Ncols) ? src[(size_t)k * Ncols + n] : 0.f;
  dst[i] = (_Float16)v;
}

// ---------------------------------------------------------------------------
// KNN: one block per (b,i). Dist to all 2048 j in LDS, 32 rounds of argmin
// with (value, lower-index) tie-break to match jax.lax.top_k stability.
__global__ __launch_bounds__(256) void k_knn(const float* __restrict__ coors,
                                             int* __restrict__ idx,
                                             float* __restrict__ reld,
                                             float* __restrict__ relc) {
  __shared__ float dist[NN];
  __shared__ float rv[256];
  __shared__ int   ri[256];
  __shared__ int   selj[KK];
  __shared__ float seld[KK];
  const int row = blockIdx.x;       // [0, B*N)
  const int b   = row >> 11;        // N = 2048
  const int t   = threadIdx.x;
  const float cx = coors[(size_t)row * 3 + 0];
  const float cy = coors[(size_t)row * 3 + 1];
  const float cz = coors[(size_t)row * 3 + 2];
  for (int j = t; j < NN; j += 256) {
    const float* cj = coors + ((size_t)(b * NN + j)) * 3;
    float dx = cx - cj[0], dy = cy - cj[1], dz = cz - cj[2];
    dist[j] = dx * dx + dy * dy + dz * dz;
  }
  __syncthreads();
  for (int k = 0; k < KK; ++k) {
    float bv = 3.4e38f;
    int   bi = 0x7fffffff;
    for (int j = t; j < NN; j += 256) {
      float v = dist[j];
      if (v < bv) { bv = v; bi = j; }
    }
    rv[t] = bv; ri[t] = bi;
    __syncthreads();
    for (int s = 128; s > 0; s >>= 1) {
      if (t < s) {
        float ov = rv[t + s]; int oi = ri[t + s];
        if (ov < rv[t] || (ov == rv[t] && oi < ri[t])) { rv[t] = ov; ri[t] = oi; }
      }
      __syncthreads();
    }
    if (t == 0) { selj[k] = ri[0]; seld[k] = rv[0]; dist[ri[0]] = 3.4e38f; }
    __syncthreads();
  }
  if (t < KK) {
    idx [(size_t)row * KK + t] = selj[t];
    reld[(size_t)row * KK + t] = seld[t];
  }
  if (t < KK * 3) {
    int k = t / 3, c = t % 3;
    int j = selj[k];
    relc[((size_t)row * KK + k) * 3 + c] =
        coors[(size_t)row * 3 + c] - coors[((size_t)(b * NN + j)) * 3 + c];
  }
}

// ---------------------------------------------------------------------------
// Fused edge pipeline: one block = 4 nodes = 128 edges. 8 waves, each owns a
// 16-row stripe. LDS: Xe[128x288] (reused as M2[128x64]) + H1[128x544]
// (reused as G[128x256]) + cw/tmp. ~210 KB of the WGP's 320 KB.
__global__ __launch_bounds__(256) void k_edge(
    const _Float16* __restrict__ feats16, const float* __restrict__ coors,
    const int* __restrict__ idx, const float* __restrict__ reld,
    const float* __restrict__ relc,
    const _Float16* __restrict__ We1p, const float* __restrict__ be1,
    const _Float16* __restrict__ We2p, const float* __restrict__ be2,
    const _Float16* __restrict__ Wc1p, const float* __restrict__ bc1,
    const float* __restrict__ Wc2, const float* __restrict__ bc2,
    float* __restrict__ m_i, float* __restrict__ coors_out) {
  extern __shared__ char smem[];
  _Float16* Xe = (_Float16*)smem;                           // [128][288]
  _Float16* H1 = (_Float16*)(smem + 128 * EINP * 2);        // [128][544]
  float*    cw = (float*)(smem + 128 * EINP * 2 + 128 * H1P * 2);  // [128]
  float*    tmp = cw + 128;                                 // [256]
  _Float16* M2 = Xe;                                        // [128][64]
  _Float16* G  = H1;                                        // [128][256]

  const int t = threadIdx.x, lane = t & 31, wave = t >> 5;
  const int gi = blockIdx.x * 4;                            // row base
  const int b  = gi >> 11;

  // Phase A: build f16 edge-input tile [feats_i | feats_j | fourier(d) | 0pad]
  {
    int e = t >> 1, half = t & 1;
    int row = gi + (e >> 5);
    int k   = e & 31;
    int j   = idx[(size_t)row * KK + k];
    const _Float16* fi = feats16 + (size_t)row * DD + half * 64;
    const _Float16* fj = feats16 + ((size_t)(b * NN + j)) * DD + half * 64;
    _Float16* xr = Xe + e * EINP;
#pragma unroll
    for (int q = 0; q < 8; ++q) {
      *(u4*)(xr + half * 64 + q * 8)      = *(const u4*)(fi + q * 8);
      *(u4*)(xr + DD + half * 64 + q * 8) = *(const u4*)(fj + q * 8);
    }
    if (half == 0) {
      float d = reld[(size_t)row * KK + k];
#pragma unroll
      for (int s = 0; s < 4; ++s) {
        float x = d / (float)(1 << s);
        xr[256 + s] = (_Float16)__sinf(x);
        xr[260 + s] = (_Float16)__cosf(x);
      }
      xr[264] = (_Float16)d;
#pragma unroll
      for (int c = 265; c < EINP; ++c) xr[c] = (_Float16)0.f;
    }
  }
  __syncthreads();

  const int m0 = wave * 16;
  const int nL = lane & 15;
  const int hiD = (lane >> 4) << 3;       // D-matrix row offset for this lane
  const int mA = m0 + nL;                 // A-matrix row for this lane

  // Phase B: edge MLP layer 1: [128x288] @ [288x544], SiLU.
  __builtin_prefetch(We1p, 0, 1);
  for (int nt = 0; nt < 34; ++nt) {
    v8f acc = {0.f, 0.f, 0.f, 0.f, 0.f, 0.f, 0.f, 0.f};
#pragma unroll
    for (int kc = 0; kc < 9; ++kc) {
      v16h a = fragA_lds(Xe, lane, mA, kc * 32, EINP);
      v16h w = fragB(We1p, lane, kc, nt, 34);
      acc = wmma_f16(a, w, acc);
    }
    int n = nt * 16 + nL;
    float bias = (n < 530) ? be1[n] : 0.f;
#pragma unroll
    for (int v = 0; v < 8; ++v)
      H1[(m0 + hiD + v) * H1P + n] = (_Float16)silu_f(acc[v] + bias);
  }
  __syncthreads();

  // Phase C: edge MLP layer 2: [128x544] @ [544x64], SiLU -> m_ij tile.
  for (int nt = 0; nt < 4; ++nt) {
    v8f acc = {0.f, 0.f, 0.f, 0.f, 0.f, 0.f, 0.f, 0.f};
#pragma unroll
    for (int kc = 0; kc < 17; ++kc) {
      v16h a = fragA_lds(H1, lane, mA, kc * 32, H1P);
      v16h w = fragB(We2p, lane, kc, nt, 4);
      acc = wmma_f16(a, w, acc);
    }
    int n = nt * 16 + nL;
    float bias = be2[n];
#pragma unroll
    for (int v = 0; v < 8; ++v)
      M2[(m0 + hiD + v) * MD + n] = (_Float16)silu_f(acc[v] + bias);
  }
  __syncthreads();

  // Phase D: coors MLP layer 1: [128x64] @ [64x256], SiLU.
  for (int nt = 0; nt < 16; ++nt) {
    v8f acc = {0.f, 0.f, 0.f, 0.f, 0.f, 0.f, 0.f, 0.f};
#pragma unroll
    for (int kc = 0; kc < 2; ++kc) {
      v16h a = fragA_lds(M2, lane, mA, kc * 32, MD);
      v16h w = fragB(Wc1p, lane, kc, nt, 16);
      acc = wmma_f16(a, w, acc);
    }
    int n = nt * 16 + nL;
    float bias = bc1[n];
#pragma unroll
    for (int v = 0; v < 8; ++v)
      G[(m0 + hiD + v) * C1 + n] = (_Float16)silu_f(acc[v] + bias);
  }
  __syncthreads();

  // Phase E: coors head: cw[e] = G[e,:] . Wc2 + bc2  (N=1, VALU dot).
  {
    int e = t >> 1, part = t & 1;
    float s = 0.f;
    const _Float16* g = G + e * C1 + part * 128;
    const float*    w = Wc2 + part * 128;
    for (int c = 0; c < 128; ++c) s += (float)g[c] * w[c];
    tmp[t] = s;
  }
  __syncthreads();
  if ((t & 1) == 0) cw[t >> 1] = tmp[t] + tmp[t + 1] + bc2[0];
  __syncthreads();

  // Phase F: coordinate update (12 outputs per block).
  if (t < 12) {
    int ii = t / 3, c = t % 3;
    int row = gi + ii;
    float s = 0.f;
    for (int k = 0; k < KK; ++k)
      s += cw[ii * KK + k] * relc[((size_t)row * KK + k) * 3 + c];
    coors_out[(size_t)row * 3 + c] = s + coors[(size_t)row * 3 + c];
  }

  // Phase G: m_i = sum_k m_ij  -> global (only edge-pipeline HBM traffic).
  {
    int ii = t >> 6, c = t & 63;
    float s = 0.f;
    for (int k = 0; k < KK; ++k) s += (float)M2[(ii * KK + k) * MD + c];
    m_i[((size_t)(gi + ii)) * MD + c] = s;
  }
}

// ---------------------------------------------------------------------------
// Node MLP: [8192x192] @ [192x256] SiLU @ [256x128] + bias + residual.
__global__ __launch_bounds__(256) void k_node(
    const float* __restrict__ feats, const _Float16* __restrict__ feats16,
    const float* __restrict__ m_i,
    const _Float16* __restrict__ Wn1p, const float* __restrict__ bn1,
    const _Float16* __restrict__ Wn2p, const float* __restrict__ bn2,
    float* __restrict__ node_out) {
  extern __shared__ char smem[];
  _Float16* Xn = (_Float16*)smem;                        // [128][192]
  _Float16* Hn = (_Float16*)(smem + 128 * NINP * 2);     // [128][256]
  const int t = threadIdx.x, lane = t & 31, wave = t >> 5;
  const int r0 = blockIdx.x * 128;
  {
    int r = t >> 1, half = t & 1;
    size_t row = r0 + r;
    if (half == 0) {
      const _Float16* f = feats16 + row * DD;
      _Float16* x = Xn + r * NINP;
#pragma unroll
      for (int q = 0; q < 16; ++q) *(u4*)(x + q * 8) = *(const u4*)(f + q * 8);
    } else {
      const float* mi = m_i + row * MD;
      _Float16* x = Xn + r * NINP + DD;
      for (int c = 0; c < MD; ++c) x[c] = (_Float16)mi[c];
    }
  }
  __syncthreads();
  const int m0 = wave * 16, nL = lane & 15, hiD = (lane >> 4) << 3;
  const int mA = m0 + nL;
  for (int nt = 0; nt < 16; ++nt) {
    v8f acc = {0.f, 0.f, 0.f, 0.f, 0.f, 0.f, 0.f, 0.f};
#pragma unroll
    for (int kc = 0; kc < 6; ++kc) {
      v16h a = fragA_lds(Xn, lane, mA, kc * 32, NINP);
      v16h w = fragB(Wn1p, lane, kc, nt, 16);
      acc = wmma_f16(a, w, acc);
    }
    int n = nt * 16 + nL;
    float bias = bn1[n];
#pragma unroll
    for (int v = 0; v < 8; ++v)
      Hn[(m0 + hiD + v) * NH + n] = (_Float16)silu_f(acc[v] + bias);
  }
  __syncthreads();
  for (int nt = 0; nt < 8; ++nt) {
    v8f acc = {0.f, 0.f, 0.f, 0.f, 0.f, 0.f, 0.f, 0.f};
#pragma unroll
    for (int kc = 0; kc < 8; ++kc) {
      v16h a = fragA_lds(Hn, lane, mA, kc * 32, NH);
      v16h w = fragB(Wn2p, lane, kc, nt, 8);
      acc = wmma_f16(a, w, acc);
    }
    int n = nt * 16 + nL;
    float bias = bn2[n];
#pragma unroll
    for (int v = 0; v < 8; ++v) {
      size_t row = r0 + m0 + hiD + v;
      node_out[row * DD + n] = acc[v] + bias + feats[row * DD + n];
    }
  }
}

// ---------------------------------------------------------------------------
extern "C" void kernel_launch(void* const* d_in, const int* in_sizes, int n_in,
                              void* d_out, int out_size, void* d_ws,
                              size_t ws_size, hipStream_t stream) {
  const float* feats = (const float*)d_in[0];   // [4,2048,128]
  const float* coors = (const float*)d_in[1];   // [4,2048,3]
  const float* We1 = (const float*)d_in[2];
  const float* be1 = (const float*)d_in[3];
  const float* We2 = (const float*)d_in[4];
  const float* be2 = (const float*)d_in[5];
  const float* Wc1 = (const float*)d_in[6];
  const float* bc1 = (const float*)d_in[7];
  const float* Wc2 = (const float*)d_in[8];
  const float* bc2 = (const float*)d_in[9];
  const float* Wn1 = (const float*)d_in[10];
  const float* bn1 = (const float*)d_in[11];
  const float* Wn2 = (const float*)d_in[12];
  const float* bn2 = (const float*)d_in[13];

  float* node_out  = (float*)d_out;
  float* coors_out = node_out + (size_t)BN * DD;

  // Workspace carve-up (~10 MB, all 16B aligned).
  char* ws = (char*)d_ws;
  size_t o = 0;
  int*      idx     = (int*)(ws + o);      o += (size_t)BN * KK * 4;
  float*    reld    = (float*)(ws + o);    o += (size_t)BN * KK * 4;
  float*    relc    = (float*)(ws + o);    o += (size_t)BN * KK * 3 * 4;
  float*    m_i     = (float*)(ws + o);    o += (size_t)BN * MD * 4;
  _Float16* feats16 = (_Float16*)(ws + o); o += (size_t)BN * DD * 2;
  _Float16* We1p    = (_Float16*)(ws + o); o += (size_t)9 * 34 * 512 * 2;
  _Float16* We2p    = (_Float16*)(ws + o); o += (size_t)17 * 4 * 512 * 2;
  _Float16* Wc1p    = (_Float16*)(ws + o); o += (size_t)2 * 16 * 512 * 2;
  _Float16* Wn1p    = (_Float16*)(ws + o); o += (size_t)6 * 16 * 512 * 2;
  _Float16* Wn2p    = (_Float16*)(ws + o); o += (size_t)8 * 8 * 512 * 2;

  // Prep
  {
    int n = BN * DD;
    k_feats16<<<(n + 255) / 256, 256, 0, stream>>>(feats, feats16, n);
  }
  {
    int tot = 9 * 34 * 512;
    k_prepack<<<(tot + 255) / 256, 256, 0, stream>>>(We1, We1p, 265, 530, 34, tot);
  }
  {
    int tot = 17 * 4 * 512;
    k_prepack<<<(tot + 255) / 256, 256, 0, stream>>>(We2, We2p, 530, 64, 4, tot);
  }
  {
    int tot = 2 * 16 * 512;
    k_prepack<<<(tot + 255) / 256, 256, 0, stream>>>(Wc1, Wc1p, 64, 256, 16, tot);
  }
  {
    int tot = 6 * 16 * 512;
    k_prepack<<<(tot + 255) / 256, 256, 0, stream>>>(Wn1, Wn1p, 192, 256, 16, tot);
  }
  {
    int tot = 8 * 8 * 512;
    k_prepack<<<(tot + 255) / 256, 256, 0, stream>>>(Wn2, Wn2p, 256, 128, 8, tot);
  }

  // KNN
  k_knn<<<BN, 256, 0, stream>>>(coors, idx, reld, relc);

  // Fused edge pipeline
  {
    size_t lds = (size_t)128 * EINP * 2 + (size_t)128 * H1P * 2 +
                 128 * 4 + 256 * 4;   // Xe + H1 + cw + tmp = 214,528 B
    k_edge<<<BN / 4, 256, lds, stream>>>(feats16, coors, idx, reld, relc,
                                         We1p, be1, We2p, be2, Wc1p, bc1,
                                         Wc2, bc2, m_i, coors_out);
  }

  // Node MLP
  {
    size_t lds = (size_t)128 * NINP * 2 + (size_t)128 * NH * 2;  // 114,688 B
    k_node<<<BN / 128, 256, lds, stream>>>(feats, feats16, m_i, Wn1p, bn1,
                                           Wn2p, bn2, node_out);
  }
}
